// TriplaneRadianceField_65661460021929
// MI455X (gfx1250) — matrix operand order repaired and checked by reading, not compiled, not run on hardware
//
#include <hip/hip_runtime.h>

typedef __attribute__((ext_vector_type(16))) _Float16 v16h;
typedef __attribute__((ext_vector_type(8)))  _Float16 v8h;
typedef __attribute__((ext_vector_type(8)))  float    v8f;
typedef __attribute__((ext_vector_type(4)))  unsigned u32x4;
typedef __attribute__((ext_vector_type(8)))  int      i32x8;
typedef __attribute__((ext_vector_type(4)))  int      i32x4;

#define NFREQ   48
#define ENC     (3*NFREQ*2)      // 288
#define FDIM    16
#define DIN     (ENC+FDIM)       // 304
#define K0PAD   320              // DIN padded to multiple of 32
#define HID     64
#define RES     32
#define WAVES   4
#define BLOCKT  (WAVES*32)       // 128 threads
#define PTSB    128              // points per block (32 per wave)

#if __has_builtin(__builtin_amdgcn_tensor_load_to_lds)
#define USE_TDM 1
#else
#define USE_TDM 0
#endif

// ---------------- WMMA fragment loaders (LDS, layouts per CDNA5 ISA 7.12.2) --
// A (16x32 f16): lane<16 holds row m=lane, K = k0+0..7 (v0..3) and k0+16..23 (v4..7)
//                lane>=16 holds row m=lane-16, K = k0+8..15 and k0+24..31
__device__ __forceinline__ v16h load_a_frag(const _Float16* rowp, int k0, int half) {
  v16h a;
  ((v8h*)&a)[0] = *(const v8h*)(rowp + k0 + half * 8);
  ((v8h*)&a)[1] = *(const v8h*)(rowp + k0 + 16 + half * 8);
  return a;
}
// B (32x16 f16): lane holds column n = lane%16; lane<16: K=k0+0..7 and k0+16..23, lane>=16: K=k0+8..15 and k0+24..31
// weights stored transposed in LDS (W^T[n][k]) so this is 2 contiguous b128 loads
__device__ __forceinline__ v16h load_b_frag(const _Float16* colp, int k0, int half) {
  v16h b;
  ((v8h*)&b)[0] = *(const v8h*)(colp + k0 + half * 8);
  ((v8h*)&b)[1] = *(const v8h*)(colp + k0 + 16 + half * 8);
  return b;
}

template<int KCH, int NT>
__device__ __forceinline__ void mlp_layer(const _Float16* arow, const _Float16* wt,
                                          int stride, int lane16, int half, v8f* acc) {
#pragma unroll
  for (int kc = 0; kc < KCH; ++kc) {
    v16h a = load_a_frag(arow, kc * 32, half);
#pragma unroll
    for (int nt = 0; nt < NT; ++nt) {
      v16h b = load_b_frag(wt + (nt * 16 + lane16) * stride, kc * 32, half);
      acc[nt] = __builtin_amdgcn_wmma_f32_16x16x32_f16(
          /*neg_a=*/false, a, /*neg_b=*/false, b,
          /*c_mod=*/(short)0, acc[nt], /*reuse_a=*/false, /*reuse_b=*/false);
    }
  }
}

// sin-activate the 16x64 D fragments and scatter into LDS activation rows (f16)
__device__ __forceinline__ void store_h_sin(const v8f* acc, _Float16* actbase,
                                            int lane16, int half) {
#pragma unroll
  for (int nt = 0; nt < 4; ++nt) {
#pragma unroll
    for (int j = 0; j < 8; ++j) {
      actbase[(j + half * 8) * K0PAD + nt * 16 + lane16] = (_Float16)__sinf(acc[nt][j]);
    }
  }
}

__device__ __forceinline__ void sample_plane(const float* __restrict__ plane,
                                             float cx, float cy, float* acc) {
  float gx = (cx + 1.0f) * 0.5f * (float)(RES - 1);
  float gy = (cy + 1.0f) * 0.5f * (float)(RES - 1);
  float x0f = fminf(fmaxf(floorf(gx), 0.0f), (float)(RES - 1));
  float y0f = fminf(fmaxf(floorf(gy), 0.0f), (float)(RES - 1));
  int x0 = (int)x0f, y0 = (int)y0f;
  int x1 = (x0 + 1 < RES) ? x0 + 1 : RES - 1;
  int y1 = (y0 + 1 < RES) ? y0 + 1 : RES - 1;
  float wx = gx - x0f, wy = gy - y0f;
  float w00 = (1.0f - wx) * (1.0f - wy), w01 = wx * (1.0f - wy);
  float w10 = (1.0f - wx) * wy,          w11 = wx * wy;
  const float4* p00 = (const float4*)(plane + (size_t)(y0 * RES + x0) * FDIM);
  const float4* p01 = (const float4*)(plane + (size_t)(y0 * RES + x1) * FDIM);
  const float4* p10 = (const float4*)(plane + (size_t)(y1 * RES + x0) * FDIM);
  const float4* p11 = (const float4*)(plane + (size_t)(y1 * RES + x1) * FDIM);
#pragma unroll
  for (int c = 0; c < 4; ++c) {
    float4 a = p00[c], b = p01[c], cte = p10[c], d = p11[c];
    acc[4 * c + 0] += w00 * a.x + w01 * b.x + w10 * cte.x + w11 * d.x;
    acc[4 * c + 1] += w00 * a.y + w01 * b.y + w10 * cte.y + w11 * d.y;
    acc[4 * c + 2] += w00 * a.z + w01 * b.z + w10 * cte.z + w11 * d.z;
    acc[4 * c + 3] += w00 * a.w + w01 * b.w + w10 * cte.w + w11 * d.w;
  }
}

__global__ void __launch_bounds__(BLOCKT)
__attribute__((amdgpu_waves_per_eu(4)))
triplane_mlp_wmma(const float* __restrict__ pos, const float* __restrict__ aabb,
                  const float* __restrict__ tp,
                  const float* __restrict__ W0, const float* __restrict__ W1,
                  const float* __restrict__ W2, const float* __restrict__ W3,
                  float* __restrict__ out, int Ntot) {
  __shared__ __attribute__((aligned(16))) _Float16 s_w0t[HID * K0PAD];   // W0^T [64][320]
  __shared__ __attribute__((aligned(16))) _Float16 s_w1t[HID * HID];     // W1^T [64][64]
  __shared__ __attribute__((aligned(16))) _Float16 s_w2t[HID * HID];     // W2^T [64][64]
  __shared__ __attribute__((aligned(16))) _Float16 s_w3t[16 * HID];      // W3^T padded [16][64]
  // act rows; ALSO overlaid as f32 TDM staging for W0 (76KB <= 80KB) before encoding
  __shared__ __attribute__((aligned(16))) _Float16 s_act[WAVES * 32 * K0PAD];
  __shared__ float s_sel[WAVES * 32];

  const int tid    = threadIdx.x;
  const int w      = tid >> 5;
  const int lane   = tid & 31;
  const int lane16 = lane & 15;
  const int half   = lane >> 4;

  const long pbase = (long)blockIdx.x * PTSB;

#if USE_TDM
  // ---- kick off the Tensor Data Mover: DMA W0 (304x64 f32) into LDS staging ----
  if (w == 0) {
    unsigned lds_stage = (unsigned)(size_t)(const void*)s_act;   // LDS addr = low 32 bits
    unsigned long long ga = (unsigned long long)(size_t)W0;
    u32x4 g0;
    g0[0] = 1u;                                                  // count=1, user mode
    g0[1] = lds_stage;                                           // lds_addr
    g0[2] = (unsigned)ga;                                        // global_addr[31:0]
    g0[3] = 0x80000000u | ((unsigned)(ga >> 32) & 0x01FFFFFFu);  // addr[56:32] | type=2
    i32x8 g1;
    g1[0] = (int)(2u << 16);       // workgroup_mask=0, data_size=2 (4B)
    g1[1] = (int)(64u << 16);      // tensor_dim0 = 64 (row of W0)
    g1[2] = (int)(304u << 16);     // tensor_dim1 = 304 rows
    g1[3] = (int)(64u << 16);      // tile_dim0 = 64
    g1[4] = 304;                   // tile_dim1 = 304
    g1[5] = 64;                    // tensor_dim0_stride = 64 elements
    g1[6] = 0;
    g1[7] = 0;
    i32x4 gz4 = {0, 0, 0, 0};      // D# groups 2/3 unused for 2D tensor
    i32x8 gz8 = {0, 0, 0, 0, 0, 0, 0, 0};
    __builtin_amdgcn_tensor_load_to_lds(g0, g1, gz4, gz4, gz8, 0);
  }
#endif

  // warm L2 path for the per-point gathers (gfx1250 global_prefetch_b8)
  __builtin_prefetch(pos + (pbase + tid) * 3, 0, 3);
  __builtin_prefetch(tp + ((tid * 257) & (3 * RES * RES * FDIM - 1)), 0, 3);

  // ---- overlapped with the TDM: stage the small weights + gather per-point data ----
  for (int i = tid; i < HID * (K0PAD - DIN); i += BLOCKT) {
    int n = i >> 4, k = DIN + (i & 15);
    s_w0t[n * K0PAD + k] = (_Float16)0.0f;   // K padding of W0^T
  }
  for (int i = tid; i < HID * HID; i += BLOCKT) {
    int k = i >> 6, n = i & 63;
    s_w1t[n * HID + k] = (_Float16)W1[i];
    s_w2t[n * HID + k] = (_Float16)W2[i];
  }
  for (int i = tid; i < 16 * HID; i += BLOCKT) {
    int k = i >> 4, n = i & 15;
    s_w3t[n * HID + k] = (n < 4) ? (_Float16)W3[k * 4 + n] : (_Float16)0.0f;
  }

  const float mn0 = aabb[0], mn1 = aabb[1], mn2 = aabb[2];
  const float iv0 = 1.0f / (aabb[3] - mn0);
  const float iv1 = 1.0f / (aabb[4] - mn1);
  const float iv2 = 1.0f / (aabb[5] - mn2);

  long p  = pbase + w * 32 + lane;
  long pc = (p < Ntot) ? p : (long)Ntot - 1;
  float px = pos[pc * 3 + 0], py = pos[pc * 3 + 1], pz = pos[pc * 3 + 2];
  float xn[3];
  xn[0] = (px - mn0) * iv0; xn[1] = (py - mn1) * iv1; xn[2] = (pz - mn2) * iv2;
  bool inbox = (xn[0] > 0.f) & (xn[0] < 1.f) & (xn[1] > 0.f) & (xn[1] < 1.f) &
               (xn[2] > 0.f) & (xn[2] < 1.f);
  float sc0 = 2.0f * xn[0] - 1.0f, sc1 = 2.0f * xn[1] - 1.0f, sc2 = 2.0f * xn[2] - 1.0f;

  float feats[FDIM];
#pragma unroll
  for (int c = 0; c < FDIM; ++c) feats[c] = 0.0f;
  sample_plane(tp + 0 * RES * RES * FDIM, sc0, sc1, feats);
  sample_plane(tp + 1 * RES * RES * FDIM, sc0, sc2, feats);
  sample_plane(tp + 2 * RES * RES * FDIM, sc1, sc2, feats);

  // ---- W0 arrives: convert f32 staging (act overlay) -> f16 transposed ----
#if USE_TDM
  if (w == 0) __builtin_amdgcn_s_wait_tensorcnt((short)0);
  __syncthreads();
  {
    const float* stage = (const float*)s_act;   // row-major [304][64] f32
    for (int i = tid; i < DIN * HID; i += BLOCKT) {
      int k = i >> 6, n = i & 63;
      s_w0t[n * K0PAD + k] = (_Float16)stage[i];
    }
  }
  __syncthreads();
#else
  for (int i = tid; i < DIN * HID; i += BLOCKT) {
    int k = i >> 6, n = i & 63;
    s_w0t[n * K0PAD + k] = (_Float16)W0[i];
  }
  __syncthreads();
#endif

  // ---- per-point encoding into act rows (one point per lane, 32 per wave) ----
  _Float16* row = s_act + (size_t)(w * 32 + lane) * K0PAD;
  const float PI_F = 3.14159265358979323846f;
#pragma unroll
  for (int d = 0; d < 3; ++d) {
    float f = xn[d] * PI_F;                      // ang_k = xn * pi * 2^k
#pragma unroll
    for (int k = 0; k < NFREQ; k += 4) {
      v8h e;
#pragma unroll
      for (int j = 0; j < 4; ++j) {
        e[2 * j]     = (_Float16)__sinf(f);
        e[2 * j + 1] = (_Float16)__cosf(f);
        f *= 2.0f;
      }
      *(v8h*)(row + d * 2 * NFREQ + 2 * k) = e;  // interleaved [sin,cos] per freq
    }
  }
  {
    v8h f0, f1, z;
#pragma unroll
    for (int c = 0; c < 8; ++c) {
      f0[c] = (_Float16)feats[c];
      f1[c] = (_Float16)feats[8 + c];
      z[c]  = (_Float16)0.0f;
    }
    *(v8h*)(row + ENC)     = f0;
    *(v8h*)(row + ENC + 8) = f1;
    *(v8h*)(row + DIN)     = z;   // K padding 304..319
    *(v8h*)(row + DIN + 8) = z;
    s_sel[w * 32 + lane] = inbox ? 1.0f : 0.0f;
  }
  // wave-local: DS ops complete in order, make rows visible to sibling lanes
  asm volatile("s_wait_dscnt 0x0" ::: "memory");

  // ---- MLP via WMMA: two 16-row M-tiles per wave ----
  for (int mt = 0; mt < 2; ++mt) {
    const _Float16* arow = s_act + (size_t)(w * 32 + mt * 16 + lane16) * K0PAD;
    _Float16* actbase    = s_act + (size_t)(w * 32 + mt * 16) * K0PAD;
    v8f zero = {};
    v8f acc[4];

    // layer 0: [16x320] x [320x64]
#pragma unroll
    for (int t = 0; t < 4; ++t) acc[t] = zero;
    mlp_layer<10, 4>(arow, s_w0t, K0PAD, lane16, half, acc);
    store_h_sin(acc, actbase, lane16, half);
    asm volatile("s_wait_dscnt 0x0" ::: "memory");

    // layer 1: [16x64] x [64x64]
#pragma unroll
    for (int t = 0; t < 4; ++t) acc[t] = zero;
    mlp_layer<2, 4>(arow, s_w1t, HID, lane16, half, acc);
    store_h_sin(acc, actbase, lane16, half);
    asm volatile("s_wait_dscnt 0x0" ::: "memory");

    // layer 2: [16x64] x [64x64]
#pragma unroll
    for (int t = 0; t < 4; ++t) acc[t] = zero;
    mlp_layer<2, 4>(arow, s_w2t, HID, lane16, half, acc);
    store_h_sin(acc, actbase, lane16, half);
    asm volatile("s_wait_dscnt 0x0" ::: "memory");

    // layer 3: [16x64] x [64x16(pad)] -> only columns 0..3 valid
    v8f o = zero;
#pragma unroll
    for (int kc = 0; kc < 2; ++kc) {
      v16h a = load_a_frag(arow, kc * 32, half);
      v16h b = load_b_frag(s_w3t + lane16 * HID, kc * 32, half);
      o = __builtin_amdgcn_wmma_f32_16x16x32_f16(false, a, false, b, (short)0, o,
                                                 false, false);
    }

    // D fragment: this lane holds column n = lane16 for rows j + half*8
    const int n = lane16;
#pragma unroll
    for (int j = 0; j < 8; ++j) {
      int rowm = j + half * 8;
      long pp = pbase + w * 32 + mt * 16 + rowm;
      if (n < 4 && pp < Ntot) {
        float v = o[j];
        if (n < 3) {
          out[pp * 3 + n] = 1.0f / (1.0f + __expf(-v));           // sigmoid rgb
        } else {
          out[(long)Ntot * 3 + pp] =
              __expf(v - 1.0f) * s_sel[w * 32 + mt * 16 + rowm];  // density
        }
      }
    }
  }
}

extern "C" void kernel_launch(void* const* d_in, const int* in_sizes, int n_in,
                              void* d_out, int out_size, void* d_ws, size_t ws_size,
                              hipStream_t stream) {
  const float* pos  = (const float*)d_in[0];
  const float* aabb = (const float*)d_in[1];
  const float* tp   = (const float*)d_in[2];
  const float* W0   = (const float*)d_in[3];
  const float* W1   = (const float*)d_in[4];
  const float* W2   = (const float*)d_in[5];
  const float* W3   = (const float*)d_in[6];
  float* out = (float*)d_out;
  int Ntot = in_sizes[0] / 3;
  int grid = (Ntot + PTSB - 1) / PTSB;
  triplane_mlp_wmma<<<grid, BLOCKT, 0, stream>>>(pos, aabb, tp, W0, W1, W2, W3, out, Ntot);
}
